// VSSBlock_69217692942427
// MI455X (gfx1250) — matrix operand order repaired and checked by reading, not compiled, not run on hardware
//
#include <hip/hip_runtime.h>
#include <hip/hip_bf16.h>

// ---------------------------------------------------------------------------
// VSSBlock for MI455X (gfx1250, wave32).  Dense projections use
// v_wmma_f32_16x16x32_bf16; the selective scan is a register-resident
// recurrence whose (dts|B|C) rows are staged into LDS with the CDNA5 async
// global->LDS DMA (double-buffered, ASYNCcnt) when available.
// cross_scan / cross_merge are index maps, never materialized.
// ---------------------------------------------------------------------------

typedef __attribute__((ext_vector_type(16))) __bf16 v16bf;
typedef __attribute__((ext_vector_type(8)))  float  v8f;

static constexpr int Bb = 16, Hh = 48, Ww = 48;
static constexpr int Ll = Hh * Ww;          // 2304
static constexpr int DM = 96, DE = 192, Nn = 16, DRr = 6, Kk = 4;
static constexpr int BL = Bb * Ll;          // 36864 rows
static constexpr int CP = 48;               // padded x_proj out channels (38->48)

#if defined(__has_builtin)
#if __has_builtin(__builtin_amdgcn_global_load_async_to_lds_b128) && \
    __has_builtin(__builtin_amdgcn_s_wait_asynccnt)
#define HAVE_ASYNC_LDS 1
#endif
#endif
#ifndef HAVE_ASYNC_LDS
#define HAVE_ASYNC_LDS 0
#endif

#if HAVE_ASYNC_LDS
// b128 async DMA operand types: int4 in global (AS1) / LDS (AS3) address space
typedef int v4i_async __attribute__((vector_size(16)));
typedef __attribute__((address_space(1))) v4i_async* async_gptr;
typedef __attribute__((address_space(3))) v4i_async* async_lptr;
#endif

// ---------------- workspace layout (bytes) ---------------------------------
static constexpr size_t OFF_WI   = 0;                                   // 384x96  bf16
static constexpr size_t OFF_WX   = OFF_WI   + (size_t)384*96*2;         // 4x48x192 bf16 (zero padded)
static constexpr size_t OFF_WO   = OFF_WX   + (size_t)4*48*192*2;       // 96x192  bf16
static constexpr size_t OFF_WE1  = OFF_WO   + (size_t)96*192*2;         // 384x192 bf16
static constexpr size_t OFF_WE3  = OFF_WE1  + (size_t)384*192*2;        // 96x384  bf16
static constexpr size_t OFF_ANEG = OFF_WE3  + (size_t)96*384*2;         // 4x192x16 f32  (-exp(A_log))
static constexpr size_t OFF_BN1  = OFF_ANEG + (size_t)4*192*16*4;       // 2x384 f32 scale/shift
static constexpr size_t OFF_BN2  = OFF_BN1  + (size_t)2*384*4;
static constexpr size_t OFF_BIG  = OFF_BN2  + (size_t)2*384*4;          // 460800, 256-aligned
static constexpr size_t SZ_XCRAW = (size_t)BL*DE*4;                     // 28.3MB
static constexpr size_t SZ_ZS    = (size_t)BL*DE*4;                     // 28.3MB
static constexpr size_t SZ_XCB   = (size_t)BL*DE*2;                     // 14.2MB
static constexpr size_t SZ_DBL   = (size_t)Bb*Kk*Ll*CP*4;               // 28.3MB
static constexpr size_t SZ_YS    = (size_t)Bb*Kk*DE*Ll*2;               // 56.6MB (== t1 fp32 bytes)
static constexpr size_t OFF_XCRAW= OFF_BIG;                             // xc_raw, later y_bf16
static constexpr size_t OFF_ZS   = OFF_XCRAW + SZ_XCRAW;                // silu(z)
static constexpr size_t OFF_XCB  = OFF_ZS    + SZ_ZS;                   // xc bf16, later t3 f32
static constexpr size_t OFF_DBL  = OFF_XCB   + SZ_XCB;                  // dbl,   later t2 bf16
static constexpr size_t OFF_YS   = OFF_DBL   + SZ_DBL;                  // ys bf16, later t1 f32
static constexpr size_t OFF_SOUT = OFF_YS    + SZ_YS;                   // 16x96 f32
// total ~156.2 MB

// ---------------- helpers --------------------------------------------------
__device__ __forceinline__ int wave_id() {
  return __builtin_amdgcn_readfirstlane((int)(threadIdx.x >> 5));   // scalar wave id
}
__device__ __forceinline__ int pos_idx(int k, int l) {
  int ll = (k & 2) ? (Ll - 1 - l) : l;          // k=2,3 are reversed streams
  if (k & 1) return (ll % Hh) * Ww + (ll / Hh); // k=1,3 traverse transposed
  return ll;
}

// ---------------- WMMA fragment loaders (wave32, 16x16x32 bf16) ------------
__device__ __forceinline__ v16bf frag_a_bf16(const __bf16* row, int lane) {
  const int k0 = (lane & 16) ? 8 : 0;
  v16bf v;
#pragma unroll
  for (int i = 0; i < 8; ++i) { v[i] = row[k0 + i]; v[i + 8] = row[k0 + 16 + i]; }
  return v;
}
__device__ __forceinline__ v16bf frag_a_f32(const float* row, int lane) {
  const int k0 = (lane & 16) ? 8 : 0;
  v16bf v;
#pragma unroll
  for (int i = 0; i < 8; ++i) {
    v[i]     = (__bf16)row[k0 + i];
    v[i + 8] = (__bf16)row[k0 + 16 + i];
  }
  return v;
}
__device__ __forceinline__ v16bf frag_b_bf16(const __bf16* wrow, int lane) {
  const int k0 = (lane & 16) ? 8 : 0;
  v16bf v;
#pragma unroll
  for (int i = 0; i < 8; ++i) { v[i] = wrow[k0 + i]; v[i + 8] = wrow[k0 + 16 + i]; }
  return v;
}
#define WMMA_BF16(a, b, c) \
  __builtin_amdgcn_wmma_f32_16x16x32_bf16(false, (a), false, (b), (short)0, (c), false, false)

// fast sigmoid: v_exp_f32 + v_rcp_f32 (no precise-div ladder)
__device__ __forceinline__ float sigf(float x) {
  return __builtin_amdgcn_rcpf(1.f + __expf(-x));
}

// ---------------- weight prep (fp32 -> bf16, BN fold, A = -exp) ------------
__global__ void k_prep(const float* __restrict__ in_w, const float* __restrict__ xp_w,
                       const float* __restrict__ out_w, const float* __restrict__ e1w,
                       const float* __restrict__ e3w,  const float* __restrict__ A_log,
                       const float* __restrict__ g1, const float* __restrict__ b1,
                       const float* __restrict__ m1, const float* __restrict__ v1,
                       const float* __restrict__ g2, const float* __restrict__ b2,
                       const float* __restrict__ m2, const float* __restrict__ v2,
                       __bf16* Wi, __bf16* Wx, __bf16* Wo, __bf16* We1, __bf16* We3,
                       float* Aneg, float* bn1, float* bn2) {
  const int tid = blockIdx.x * blockDim.x + threadIdx.x;
  const int nt  = gridDim.x * blockDim.x;
  for (int i = tid; i < 384 * 96; i += nt)  Wi[i]  = (__bf16)in_w[i];
  for (int i = tid; i < 4 * CP * 192; i += nt) {
    int d = i % 192, c = (i / 192) % CP, k = i / (192 * CP);
    Wx[i] = (c < DRr + 2 * Nn) ? (__bf16)xp_w[((size_t)k * 38 + c) * 192 + d]
                               : (__bf16)0.0f;
  }
  for (int i = tid; i < 96 * 192; i += nt)  Wo[i]  = (__bf16)out_w[i];
  for (int i = tid; i < 384 * 192; i += nt) We1[i] = (__bf16)e1w[i];
  for (int i = tid; i < 96 * 384; i += nt)  We3[i] = (__bf16)e3w[i];
  for (int i = tid; i < 4 * 192 * 16; i += nt) Aneg[i] = -__expf(A_log[i]);
  for (int i = tid; i < 384; i += nt) {
    float s1 = g1[i] * rsqrtf(v1[i] + 1e-5f);
    bn1[i] = s1; bn1[384 + i] = b1[i] - m1[i] * s1;
    float s2 = g2[i] * rsqrtf(v2[i] + 1e-5f);
    bn2[i] = s2; bn2[384 + i] = b2[i] - m2[i] * s2;
  }
}

// ---------------- in_proj: xz = x @ Wi^T ; split into xc_raw / silu(z) -----
__global__ void __launch_bounds__(256) k_inproj(const float* __restrict__ x,
    const __bf16* __restrict__ Wi, float* __restrict__ xc_raw,
    float* __restrict__ z_silu) {
  const int lane = threadIdx.x & 31;
  const int wv   = wave_id();                     // scalar -> scalar ct loop
  const int rt   = blockIdx.x;                    // 16-row tile over B*L
  const float* arow = x + (size_t)(rt * 16 + (lane & 15)) * DM;
  for (int ct = wv; ct < 24; ct += 8) {           // 24 col tiles of 16 (=384)
    const __bf16* wrow = Wi + (size_t)(ct * 16 + (lane & 15)) * DM;
    v8f acc = {};
#pragma unroll
    for (int kk = 0; kk < DM; kk += 32)
      acc = WMMA_BF16(frag_a_f32(arow + kk, lane), frag_b_bf16(wrow + kk, lane), acc);
    const int n = lane & 15, mb = (lane & 16) ? 8 : 0;
    if (ct < 12) {                                // uniform: xc half (e<192)
      const int e = ct * 16 + n;
#pragma unroll
      for (int r = 0; r < 8; ++r)
        xc_raw[((size_t)rt * 16 + mb + r) * DE + e] = acc[r];
    } else {                                      // uniform: z half, fuse SiLU
      const int e = ct * 16 + n - DE;
#pragma unroll
      for (int r = 0; r < 8; ++r) {
        float v = acc[r];
        z_silu[((size_t)rt * 16 + mb + r) * DE + e] = v * sigf(v);
      }
    }
  }
}

// ---------------- depthwise 3x3 conv + bias + SiLU -> bf16 -----------------
__global__ void k_dwconv(const float* __restrict__ xc_raw,
                         const float* __restrict__ cw, const float* __restrict__ cb,
                         __bf16* __restrict__ xc_bf) {
  const size_t idx = (size_t)blockIdx.x * blockDim.x + threadIdx.x;
  if (idx >= (size_t)BL * DE) return;
  const int c = idx % DE;
  const int w = (idx / DE) % Ww;
  const int h = (idx / ((size_t)DE * Ww)) % Hh;
  const int b = idx / ((size_t)DE * Ww * Hh);
  float s = cb[c];
#pragma unroll
  for (int kh = 0; kh < 3; ++kh) {
    const int hy = h + kh - 1; if (hy < 0 || hy >= Hh) continue;
#pragma unroll
    for (int kw = 0; kw < 3; ++kw) {
      const int wx = w + kw - 1; if (wx < 0 || wx >= Ww) continue;
      s += xc_raw[(((size_t)b * Hh + hy) * Ww + wx) * DE + c] * cw[(kh * 3 + kw) * DE + c];
    }
  }
  s = s * sigf(s);
  xc_bf[idx] = (__bf16)s;
}

// ---------------- x_proj over 4 gathered directions (out padded to 48) -----
__global__ void __launch_bounds__(128) k_xproj(const __bf16* __restrict__ xc_bf,
    const __bf16* __restrict__ Wx, float* __restrict__ dbl) {
  const int lane = threadIdx.x & 31;
  const int t  = blockIdx.x * 4 + wave_id();              // 9216 wave-tiles
  const int lt = t % (Ll / 16);
  const int k  = (t / (Ll / 16)) % Kk;
  const int b  = t / ((Ll / 16) * Kk);
  const int p  = pos_idx(k, lt * 16 + (lane & 15));       // gathered A row
  const __bf16* arow = xc_bf + ((size_t)b * Ll + p) * DE;
  float* obase = dbl + (size_t)(b * Kk + k) * Ll * CP;
  for (int ct = 0; ct < 3; ++ct) {                        // 48 cols
    const __bf16* wrow = Wx + (size_t)(k * CP + ct * 16 + (lane & 15)) * DE;
    v8f acc = {};
#pragma unroll
    for (int kk = 0; kk < DE; kk += 32)
      acc = WMMA_BF16(frag_a_bf16(arow + kk, lane), frag_b_bf16(wrow + kk, lane), acc);
    const int n = lane & 15, mb = (lane & 16) ? 8 : 0;
#pragma unroll
    for (int r = 0; r < 8; ++r)
      obase[(size_t)(lt * 16 + mb + r) * CP + ct * 16 + n] = acc[r];
  }
}

// ---------------- selective scan (fused dt_proj + softplus + D skip) -------
// block = (b,k); thread = channel d; h[16] in VGPRs.  The (dts|B|C) rows are
// double-buffered into LDS via async global->LDS DMA (ASYNCcnt) when the
// toolchain exposes it; otherwise synchronous staging + global prefetch.
__global__ void __launch_bounds__(192) k_scan(const float* __restrict__ dbl,
    const __bf16* __restrict__ xc_bf, const float* __restrict__ Aneg,
    const float* __restrict__ dtw_g, const float* __restrict__ dtb_g,
    const float* __restrict__ Ds_g, __bf16* __restrict__ ys) {
  __shared__ __align__(16) float chunk[2][48 * CP];   // 2 x 9216 B
  const int d  = threadIdx.x;
  const int bk = blockIdx.x;
  const int b  = bk / Kk, k = bk % Kk;
  const float* dblp = dbl + (size_t)bk * Ll * CP;
  float An[16], Wt[6], h[16];
#pragma unroll
  for (int n = 0; n < 16; ++n) { An[n] = Aneg[((size_t)k * DE + d) * 16 + n]; h[n] = 0.f; }
#pragma unroll
  for (int r = 0; r < 6; ++r) Wt[r] = dtw_g[((size_t)k * DE + d) * 6 + r];
  const float dtb = dtb_g[k * DE + d];
  const float Dd  = Ds_g[k * DE + d];
  __bf16* ysp = ys + ((size_t)bk * DE + d) * Ll;

#if HAVE_ASYNC_LDS
  auto issue_chunk = [&](float* dst, int c0base) {
    char* gsrc = (char*)(dblp + (size_t)c0base * CP);
    char* ldst = (char*)dst;
#pragma unroll
    for (int it = 0; it < 3; ++it)                 // 192 lanes x 16B x 3 = 9216B
      __builtin_amdgcn_global_load_async_to_lds_b128(
          (async_gptr)(gsrc + it * 3072 + (int)threadIdx.x * 16),
          (async_lptr)(ldst + it * 3072 + (int)threadIdx.x * 16),
          0, 0);
  };
  issue_chunk(chunk[0], 0);
#endif

  for (int c0 = 0; c0 < Ll; c0 += 48) {
    const int cur = (c0 / 48) & 1;
#if HAVE_ASYNC_LDS
    __builtin_amdgcn_s_wait_asynccnt(0);           // this wave's DMA landed
    __syncthreads();                               // all waves' DMA landed,
                                                   // prev buffer fully read
    if (c0 + 48 < Ll) issue_chunk(chunk[cur ^ 1], c0 + 48);
#else
    __syncthreads();
    for (int i = threadIdx.x; i < 48 * CP; i += 192)
      chunk[cur][i] = dblp[(size_t)c0 * CP + i];
    __syncthreads();
    if (c0 + 48 < Ll)                              // global_prefetch next chunk
      __builtin_prefetch(dblp + (size_t)(c0 + 48) * CP + threadIdx.x * 12, 0, 0);
#endif
    const float* cbase = chunk[cur];
    for (int li = 0; li < 48; ++li) {
      const int l = c0 + li;
      const float* row = cbase + li * CP;          // [0:6)=dts [6:22)=B [22:38)=C
      float da = dtb;
#pragma unroll
      for (int r = 0; r < 6; ++r) da += row[r] * Wt[r];
      const float delta = (da > 20.f) ? da : __logf(1.f + __expf(da));
      const float u  = (float)xc_bf[((size_t)b * Ll + pos_idx(k, l)) * DE + d];
      const float du = delta * u;
      float y = Dd * u;
#pragma unroll
      for (int n = 0; n < 16; ++n) {
        h[n] = h[n] * __expf(delta * An[n]) + du * row[6 + n];
        y += h[n] * row[22 + n];
      }
      ysp[l] = (__bf16)y;
    }
  }
}

// ---------------- cross_merge + gate with silu(z) -> y bf16 ----------------
__global__ void k_merge(const __bf16* __restrict__ ys, const float* __restrict__ z_silu,
                        __bf16* __restrict__ y_bf) {
  const size_t idx = (size_t)blockIdx.x * blockDim.x + threadIdx.x;
  if (idx >= (size_t)BL * DE) return;
  const int dd = idx % DE;
  const int l  = (idx / DE) % Ll;
  const int b  = idx / ((size_t)DE * Ll);
  const int lp = (l % Ww) * Hh + (l / Ww);          // transposed-stream index
  const size_t base = ((size_t)b * Kk) * DE + dd;
  const float v =
      (float)ys[(base + 0 * DE) * Ll + l] +
      (float)ys[(base + 2 * DE) * Ll + (Ll - 1 - l)] +
      (float)ys[(base + 1 * DE) * Ll + lp] +
      (float)ys[(base + 3 * DE) * Ll + (Ll - 1 - lp)];
  y_bf[idx] = (__bf16)(v * z_silu[idx]);
}

// ---------------- out_proj: out = y @ Wo^T ---------------------------------
__global__ void __launch_bounds__(192) k_outproj(const __bf16* __restrict__ y_bf,
    const __bf16* __restrict__ Wo, float* __restrict__ out) {
  const int lane = threadIdx.x & 31, ct = wave_id();          // 6 col tiles
  const int rt = blockIdx.x;
  const __bf16* arow = y_bf + (size_t)(rt * 16 + (lane & 15)) * DE;
  const __bf16* wrow = Wo   + (size_t)(ct * 16 + (lane & 15)) * DE;
  v8f acc = {};
#pragma unroll
  for (int kk = 0; kk < DE; kk += 32)
    acc = WMMA_BF16(frag_a_bf16(arow + kk, lane), frag_b_bf16(wrow + kk, lane), acc);
  const int n = lane & 15, mb = (lane & 16) ? 8 : 0;
#pragma unroll
  for (int r = 0; r < 8; ++r)
    out[(size_t)(rt * 16 + mb + r) * DM + ct * 16 + n] = acc[r];
}

// ---------------- e1 + BN1 + ReLU ------------------------------------------
__global__ void __launch_bounds__(256) k_e1(const __bf16* __restrict__ y_bf,
    const __bf16* __restrict__ We1, const float* __restrict__ e1_b,
    const float* __restrict__ bn1, float* __restrict__ t1) {
  const int lane = threadIdx.x & 31, wv = wave_id();
  const int rt = blockIdx.x;
  const __bf16* arow = y_bf + (size_t)(rt * 16 + (lane & 15)) * DE;
  for (int ct = wv; ct < 24; ct += 8) {
    const __bf16* wrow = We1 + (size_t)(ct * 16 + (lane & 15)) * DE;
    v8f acc = {};
#pragma unroll
    for (int kk = 0; kk < DE; kk += 32)
      acc = WMMA_BF16(frag_a_bf16(arow + kk, lane), frag_b_bf16(wrow + kk, lane), acc);
    const int n = lane & 15, mb = (lane & 16) ? 8 : 0;
    const int c = ct * 16 + n;
    const float sc = bn1[c], sh = bn1[384 + c], bb = e1_b[c];
#pragma unroll
    for (int r = 0; r < 8; ++r) {
      float v = (acc[r] + bb) * sc + sh;
      t1[((size_t)rt * 16 + mb + r) * 384 + c] = fmaxf(v, 0.f);
    }
  }
}

// ---------------- e2 depthwise conv + BN2 + ReLU -> bf16 -------------------
__global__ void k_e2(const float* __restrict__ t1, const float* __restrict__ w2,
                     const float* __restrict__ b2, const float* __restrict__ bn2,
                     __bf16* __restrict__ t2) {
  const size_t idx = (size_t)blockIdx.x * blockDim.x + threadIdx.x;
  if (idx >= (size_t)BL * 384) return;
  const int c = idx % 384;
  const int w = (idx / 384) % Ww;
  const int h = (idx / ((size_t)384 * Ww)) % Hh;
  const int b = idx / ((size_t)384 * Ww * Hh);
  float s = b2[c];
#pragma unroll
  for (int kh = 0; kh < 3; ++kh) {
    const int hy = h + kh - 1; if (hy < 0 || hy >= Hh) continue;
#pragma unroll
    for (int kw = 0; kw < 3; ++kw) {
      const int wx = w + kw - 1; if (wx < 0 || wx >= Ww) continue;
      s += t1[(((size_t)b * Hh + hy) * Ww + wx) * 384 + c] * w2[(kh * 3 + kw) * 384 + c];
    }
  }
  s = s * bn2[c] + bn2[384 + c];
  t2[idx] = (__bf16)fmaxf(s, 0.f);
}

// ---------------- e3: t3 = t2 @ We3^T + b ----------------------------------
__global__ void __launch_bounds__(192) k_e3(const __bf16* __restrict__ t2,
    const __bf16* __restrict__ We3, const float* __restrict__ e3_b,
    float* __restrict__ t3) {
  const int lane = threadIdx.x & 31, ct = wave_id();          // 6 col tiles
  const int rt = blockIdx.x;
  const __bf16* arow = t2  + (size_t)(rt * 16 + (lane & 15)) * 384;
  const __bf16* wrow = We3 + (size_t)(ct * 16 + (lane & 15)) * 384;
  v8f acc = {};
#pragma unroll
  for (int kk = 0; kk < 384; kk += 32)
    acc = WMMA_BF16(frag_a_bf16(arow + kk, lane), frag_b_bf16(wrow + kk, lane), acc);
  const int n = lane & 15, mb = (lane & 16) ? 8 : 0;
  const int c = ct * 16 + n;
  const float bb = e3_b[c];
#pragma unroll
  for (int r = 0; r < 8; ++r)
    t3[((size_t)rt * 16 + mb + r) * DM + c] = acc[r] + bb;
}

// ---------------- SE branch: mean -> fc -> relu -> fc -> sigmoid -----------
__global__ void __launch_bounds__(128) k_se(const float* __restrict__ t3,
    const float* __restrict__ w1, const float* __restrict__ b1,
    const float* __restrict__ w2, const float* __restrict__ b2,
    float* __restrict__ s_out) {
  __shared__ float sbuf[96], hbuf[48];
  const int b = blockIdx.x, tid = threadIdx.x;
  if (tid < 96) {
    float s = 0.f;
    for (int l = 0; l < Ll; ++l) s += t3[((size_t)b * Ll + l) * DM + tid];
    sbuf[tid] = s * (1.f / Ll);
  }
  __syncthreads();
  if (tid < 48) {
    float a = b1[tid];
#pragma unroll
    for (int c = 0; c < 96; ++c) a += sbuf[c] * w1[tid * 96 + c];
    hbuf[tid] = fmaxf(a, 0.f);
  }
  __syncthreads();
  if (tid < 96) {
    float a = b2[tid];
#pragma unroll
    for (int j = 0; j < 48; ++j) a += hbuf[j] * w2[tid * 48 + j];
    s_out[b * DM + tid] = sigf(a);
  }
}

// ---------------- final: out += t3 * s -------------------------------------
__global__ void k_final(const float* __restrict__ t3, const float* __restrict__ s,
                        float* __restrict__ out) {
  const size_t idx = (size_t)blockIdx.x * blockDim.x + threadIdx.x;
  if (idx >= (size_t)BL * DM) return;
  const int c = idx % DM;
  const int b = idx / ((size_t)Ll * DM);
  out[idx] += t3[idx] * s[b * DM + c];
}

// ---------------------------------------------------------------------------
extern "C" void kernel_launch(void* const* d_in, const int* in_sizes, int n_in,
                              void* d_out, int out_size, void* d_ws, size_t ws_size,
                              hipStream_t stream) {
  (void)in_sizes; (void)n_in; (void)out_size; (void)ws_size;
  const float* x         = (const float*)d_in[0];
  const float* in_proj_w = (const float*)d_in[1];
  const float* conv_w    = (const float*)d_in[2];
  const float* conv_b    = (const float*)d_in[3];
  const float* x_proj_w  = (const float*)d_in[4];
  const float* dt_proj_w = (const float*)d_in[5];
  const float* dt_proj_b = (const float*)d_in[6];
  const float* A_log     = (const float*)d_in[7];
  const float* Ds        = (const float*)d_in[8];
  const float* out_w     = (const float*)d_in[9];
  const float* e1_w      = (const float*)d_in[10];
  const float* e1_b      = (const float*)d_in[11];
  const float* bn1_g     = (const float*)d_in[12];
  const float* bn1_b     = (const float*)d_in[13];
  const float* bn1_m     = (const float*)d_in[14];
  const float* bn1_v     = (const float*)d_in[15];
  const float* e2_w      = (const float*)d_in[16];
  const float* e2_b      = (const float*)d_in[17];
  const float* bn2_g     = (const float*)d_in[18];
  const float* bn2_b     = (const float*)d_in[19];
  const float* bn2_m     = (const float*)d_in[20];
  const float* bn2_v     = (const float*)d_in[21];
  const float* e3_w      = (const float*)d_in[22];
  const float* e3_b      = (const float*)d_in[23];
  const float* se1_w     = (const float*)d_in[24];
  const float* se1_b     = (const float*)d_in[25];
  const float* se2_w     = (const float*)d_in[26];
  const float* se2_b     = (const float*)d_in[27];

  char* ws = (char*)d_ws;
  __bf16* Wi    = (__bf16*)(ws + OFF_WI);
  __bf16* Wx    = (__bf16*)(ws + OFF_WX);
  __bf16* Wo    = (__bf16*)(ws + OFF_WO);
  __bf16* We1   = (__bf16*)(ws + OFF_WE1);
  __bf16* We3   = (__bf16*)(ws + OFF_WE3);
  float*  Aneg  = (float*) (ws + OFF_ANEG);
  float*  bn1   = (float*) (ws + OFF_BN1);
  float*  bn2   = (float*) (ws + OFF_BN2);
  float*  xcraw = (float*) (ws + OFF_XCRAW);
  float*  zsilu = (float*) (ws + OFF_ZS);
  __bf16* xc_bf = (__bf16*)(ws + OFF_XCB);
  float*  dbl   = (float*) (ws + OFF_DBL);
  __bf16* ys    = (__bf16*)(ws + OFF_YS);
  float*  sout  = (float*) (ws + OFF_SOUT);
  // lifetime-based aliases (safe: separated by kernel order on the stream)
  __bf16* y_bf  = (__bf16*)(ws + OFF_XCRAW);   // over xc_raw (dead after dwconv)
  float*  t1    = (float*) (ws + OFF_YS);      // over ys     (dead after merge)
  __bf16* t2    = (__bf16*)(ws + OFF_DBL);     // over dbl    (dead after scan)
  float*  t3    = (float*) (ws + OFF_XCB);     // over xc_bf  (dead after scan)
  float*  out   = (float*)d_out;

  k_prep<<<64, 256, 0, stream>>>(in_proj_w, x_proj_w, out_w, e1_w, e3_w, A_log,
                                 bn1_g, bn1_b, bn1_m, bn1_v, bn2_g, bn2_b, bn2_m, bn2_v,
                                 Wi, Wx, Wo, We1, We3, Aneg, bn1, bn2);
  k_inproj <<<BL / 16, 256, 0, stream>>>(x, Wi, xcraw, zsilu);
  k_dwconv <<<((size_t)BL * DE + 255) / 256, 256, 0, stream>>>(xcraw, conv_w, conv_b, xc_bf);
  k_xproj  <<<Bb * Kk * (Ll / 16) / 4, 128, 0, stream>>>(xc_bf, Wx, dbl);
  k_scan   <<<Bb * Kk, DE, 0, stream>>>(dbl, xc_bf, Aneg, dt_proj_w, dt_proj_b, Ds, ys);
  k_merge  <<<((size_t)BL * DE + 255) / 256, 256, 0, stream>>>(ys, zsilu, y_bf);
  k_outproj<<<BL / 16, 192, 0, stream>>>(y_bf, Wo, out);
  k_e1     <<<BL / 16, 256, 0, stream>>>(y_bf, We1, e1_b, bn1, t1);
  k_e2     <<<((size_t)BL * 384 + 255) / 256, 256, 0, stream>>>(t1, e2_w, e2_b, bn2, t2);
  k_e3     <<<BL / 16, 192, 0, stream>>>(t2, We3, e3_b, t3);
  k_se     <<<Bb, 128, 0, stream>>>(t3, se1_w, se1_b, se2_w, se2_b, sout);
  k_final  <<<((size_t)BL * DM + 255) / 256, 256, 0, stream>>>(t3, sout, out);
}
